// MultiHeadAttention_28028956574007
// MI455X (gfx1250) — matrix operand order repaired and checked by reading, not compiled, not run on hardware
//
#include <hip/hip_runtime.h>
#include <stdint.h>

// ---------- types ----------
typedef __attribute__((ext_vector_type(16))) __bf16 v16bf;
typedef __attribute__((ext_vector_type(8)))  float  v8f;
typedef __attribute__((ext_vector_type(4)))  unsigned int u32x4;
typedef __attribute__((ext_vector_type(4)))  int i4v;

union Frag { u32x4 u[2]; v16bf v; };

// ---------- helpers ----------
__device__ __forceinline__ unsigned short f2bf(float f) {
    union { float f; unsigned u; } c; c.f = f;
    unsigned u = c.u;
    unsigned r = u + 0x7fffu + ((u >> 16) & 1u);   // round-to-nearest-even
    return (unsigned short)(r >> 16);
}
__device__ __forceinline__ float bf2f(unsigned short h) {
    union { unsigned u; float f; } c; c.u = ((unsigned)h) << 16;
    return c.f;
}

// ---------- CDNA5 async global->LDS path (guarded; falls back to VGPR copy) ----------
#if __has_builtin(__builtin_amdgcn_global_load_async_to_lds_b128) && \
    __has_builtin(__builtin_amdgcn_s_wait_asynccnt)
#define HAVE_ASYNC_LDS 1
#endif

__device__ __forceinline__ void async_cp16(const unsigned short* g, unsigned short* l) {
#ifdef HAVE_ASYNC_LDS
    __builtin_amdgcn_global_load_async_to_lds_b128(
        (__attribute__((address_space(1))) i4v*)g,
        (__attribute__((address_space(3))) i4v*)l, 0, 0);
#else
    *(u32x4*)l = *(const u32x4*)g;
#endif
}
__device__ __forceinline__ void wait_stage() {
#ifdef HAVE_ASYNC_LDS
    __builtin_amdgcn_s_wait_asynccnt(0);
#endif
}

// ---------- conversion kernels ----------
__global__ void cvt_f32_bf16(const float* __restrict__ src,
                             unsigned short* __restrict__ dst, int n) {
    int i = blockIdx.x * blockDim.x + threadIdx.x;
    if (i < n) dst[i] = f2bf(src[i]);
}

// Wq[n][k] scaled by scale[n % 64]  (folds q*scale into the weight)
__global__ void cvt_wq_scaled(const float* __restrict__ W,
                              const float* __restrict__ scale,
                              unsigned short* __restrict__ dst, int n) {
    int i = blockIdx.x * blockDim.x + threadIdx.x;
    if (i < n) dst[i] = f2bf(W[i] * scale[(i >> 10) & 63]);
}

__global__ void scale_bias(const float* __restrict__ b,
                           const float* __restrict__ scale,
                           float* __restrict__ out, int n) {
    int i = blockIdx.x * blockDim.x + threadIdx.x;
    if (i < n) out[i] = b[i] * scale[i & 63];
}

// ---------- bf16 WMMA GEMM: C[M,N] = A[M,K] * W[N,K]^T + bias ----------
#define BM 128
#define BN 256
#define BK 32
#define LDA 40   // padded LDS row stride (elems) -> 80B = 20-bank stride, conflict-free
#define LDB 40
#define ASZ (BM * LDA)   // elems per A buffer
#define BSZ (BN * LDB)   // elems per B buffer

// stage one 128x32 A tile + 256x32 B tile into LDS buffers (256 threads)
__device__ __forceinline__ void stage_tiles(const unsigned short* __restrict__ Ab,
                                            const unsigned short* __restrict__ Wb,
                                            unsigned short* sAbuf,
                                            unsigned short* sBbuf,
                                            int tid, int K)
{
    const int r    = tid >> 1;   // 0..127
    const int half = tid & 1;
    const unsigned short* ga = Ab + (size_t)r * K + half * 16;
    unsigned short*       la = sAbuf + r * LDA + half * 16;
    async_cp16(ga,     la);
    async_cp16(ga + 8, la + 8);

    const unsigned short* gb = Wb + (size_t)tid * K;   // row n = tid (0..255)
    unsigned short*       lb = sBbuf + tid * LDB;
#pragma unroll
    for (int c = 0; c < 4; ++c) async_cp16(gb + c * 8, lb + c * 8);
}

__global__ __launch_bounds__(256) void gemm_bf16_nt(
    const unsigned short* __restrict__ A,   // [M][K] bf16
    const unsigned short* __restrict__ W,   // [N][K] bf16 (torch Linear layout)
    const float* __restrict__ bias,         // [N] fp32
    void* __restrict__ Cout,
    int M, int N, int K, int store_bf16)
{
    __shared__ __align__(16) unsigned short sA[2 * ASZ];
    __shared__ __align__(16) unsigned short sB[2 * BSZ];

    const int tid   = threadIdx.x;
    const int lane  = tid & 31;
    const int wave  = tid >> 5;      // 0..7
    const int waveM = wave >> 2;     // 0..1  (64 rows each)
    const int waveN = wave & 3;      // 0..3  (64 cols each)

    const int m0 = blockIdx.y * BM;
    const int n0 = blockIdx.x * BN;

    const int lrow = lane & 15;
    const int kh   = lane >> 4;

    v8f acc[4][4];
    const v8f vzero = {0.f,0.f,0.f,0.f,0.f,0.f,0.f,0.f};
#pragma unroll
    for (int mt = 0; mt < 4; ++mt)
#pragma unroll
        for (int nt = 0; nt < 4; ++nt) acc[mt][nt] = vzero;

    const unsigned short* Abase = A + (size_t)m0 * K;
    const unsigned short* Wbase = W + (size_t)n0 * K;

    // prologue: stage first k-tile into buffer 0
    stage_tiles(Abase, Wbase, sA, sB, tid, K);

    const int nsteps = K / BK;
    for (int s = 0; s < nsteps; ++s) {
        const int p = s & 1;
        wait_stage();          // this wave's async copies into buf p complete
        __syncthreads();       // everyone's copies visible; prior reads of buf p done

        if (s + 1 < nsteps)
            stage_tiles(Abase + (s + 1) * BK, Wbase + (s + 1) * BK,
                        sA + (p ^ 1) * ASZ, sB + (p ^ 1) * BSZ, tid, K);

        // ---- build fragments per ISA VGPR layouts ----
        const unsigned short* sAp = sA + p * ASZ;
        const unsigned short* sBp = sB + p * BSZ;
        Frag a[4], b[4];
#pragma unroll
        for (int mt = 0; mt < 4; ++mt) {
            const unsigned short* base = &sAp[(waveM * 64 + mt * 16 + lrow) * LDA];
            a[mt].u[0] = *(const u32x4*)(base + kh * 8);        // K = 8*kh .. +7
            a[mt].u[1] = *(const u32x4*)(base + 16 + kh * 8);   // K = 16+8*kh .. +7
        }
#pragma unroll
        for (int nt = 0; nt < 4; ++nt) {
            const unsigned short* base = &sBp[(waveN * 64 + nt * 16 + lrow) * LDB];
            b[nt].u[0] = *(const u32x4*)(base + kh * 8);        // K = 8*kh .. +7
            b[nt].u[1] = *(const u32x4*)(base + 16 + kh * 8);   // K = 16+8*kh .. +7
        }

#pragma unroll
        for (int mt = 0; mt < 4; ++mt)
#pragma unroll
            for (int nt = 0; nt < 4; ++nt)
                acc[mt][nt] = __builtin_amdgcn_wmma_f32_16x16x32_bf16(
                    false, a[mt].v, false, b[nt].v,
                    (short)0, acc[mt][nt], false, false);

        __syncthreads();       // reads of buf p done before it is re-staged
    }

    // ---- epilogue: C/D layout -> element (m = base + i + 8*kh, n = lrow) ----
#pragma unroll
    for (int mt = 0; mt < 4; ++mt) {
#pragma unroll
        for (int nt = 0; nt < 4; ++nt) {
            const int n = n0 + waveN * 64 + nt * 16 + lrow;
            const float bb = bias ? bias[n] : 0.f;
            const int mbase = m0 + waveM * 64 + mt * 16 + kh * 8;
#pragma unroll
            for (int i = 0; i < 8; ++i) {
                const float val = acc[mt][nt][i] + bb;
                const size_t idx = (size_t)(mbase + i) * N + n;
                if (store_bf16) ((unsigned short*)Cout)[idx] = f2bf(val);
                else            ((float*)Cout)[idx]          = val;
            }
        }
    }
}

// ---------- token-local attention (scores are [16 heads x 16 heads] per token) ----------
#define TOK_PER_BLK 8
__global__ __launch_bounds__(128) void attn_kernel(
    const unsigned short* __restrict__ qb,   // [M][1024] bf16 (scale pre-folded)
    const unsigned short* __restrict__ kb,
    const unsigned short* __restrict__ vb,
    unsigned short* __restrict__ ctx)
{
    __shared__ __align__(16) unsigned short sk[TOK_PER_BLK * 1024];
    __shared__ __align__(16) unsigned short sv[TOK_PER_BLK * 1024];

    const int tid = threadIdx.x;
    const size_t tok0 = (size_t)blockIdx.x * TOK_PER_BLK;

    // stage k, v for 8 tokens (16KB each, contiguous)
    const u32x4* gk = (const u32x4*)(kb + tok0 * 1024);
    const u32x4* gv = (const u32x4*)(vb + tok0 * 1024);
    u32x4* lk = (u32x4*)sk;
    u32x4* lv = (u32x4*)sv;
#pragma unroll
    for (int i = 0; i < 8; ++i) {
        lk[tid + i * 128] = gk[tid + i * 128];
        lv[tid + i * 128] = gv[tid + i * 128];
    }
    __syncthreads();

    const int t = tid >> 4;   // token within block
    const int h = tid & 15;   // head

    const unsigned short* qrow = qb + (tok0 + t) * 1024 + h * 64;
    float q[64];
#pragma unroll
    for (int d = 0; d < 64; ++d) q[d] = bf2f(qrow[d]);

    float sc[16];
    const unsigned short* kbase = sk + t * 1024;
#pragma unroll
    for (int tt = 0; tt < 16; ++tt) {
        float s = 0.f;
#pragma unroll
        for (int d = 0; d < 64; ++d) s += q[d] * bf2f(kbase[tt * 64 + d]);
        sc[tt] = s * 0.125f;   // 1/sqrt(64)
    }
    float mx = sc[0];
#pragma unroll
    for (int tt = 1; tt < 16; ++tt) mx = fmaxf(mx, sc[tt]);
    float sum = 0.f;
#pragma unroll
    for (int tt = 0; tt < 16; ++tt) { sc[tt] = __expf(sc[tt] - mx); sum += sc[tt]; }
    const float inv = 1.f / sum;

    float o[64];
#pragma unroll
    for (int d = 0; d < 64; ++d) o[d] = 0.f;
    const unsigned short* vbase = sv + t * 1024;
#pragma unroll
    for (int tt = 0; tt < 16; ++tt) {
        const float w = sc[tt] * inv;
        const unsigned short* vrow = vbase + tt * 64;
#pragma unroll
        for (int d = 0; d < 64; ++d) o[d] += w * bf2f(vrow[d]);
    }

    unsigned short* orow = ctx + (tok0 + t) * 1024 + h * 64;
#pragma unroll
    for (int d = 0; d < 64; ++d) orow[d] = f2bf(o[d]);
}

// ---------- host ----------
extern "C" void kernel_launch(void* const* d_in, const int* in_sizes, int n_in,
                              void* d_out, int out_size, void* d_ws, size_t ws_size,
                              hipStream_t stream) {
    const float* x     = (const float*)d_in[0];
    const float* Wq    = (const float*)d_in[1];
    const float* bq    = (const float*)d_in[2];
    const float* Wk    = (const float*)d_in[3];
    const float* bk    = (const float*)d_in[4];
    const float* Wv    = (const float*)d_in[5];
    const float* bv    = (const float*)d_in[6];
    const float* Wo    = (const float*)d_in[7];
    const float* bo    = (const float*)d_in[8];
    const float* scale = (const float*)d_in[9];

    const int M = 4 * 4096;   // 16384 tokens
    const int H = 1024;

    size_t off = 0;
    auto alloc = [&](size_t bytes) -> void* {
        void* p = (char*)d_ws + off;
        off += (bytes + 255) & ~(size_t)255;
        return p;
    };
    unsigned short* xb   = (unsigned short*)alloc((size_t)M * H * 2);
    unsigned short* wqb  = (unsigned short*)alloc((size_t)H * H * 2);
    unsigned short* wkb  = (unsigned short*)alloc((size_t)H * H * 2);
    unsigned short* wvb  = (unsigned short*)alloc((size_t)H * H * 2);
    unsigned short* wob  = (unsigned short*)alloc((size_t)H * H * 2);
    float*          bqs  = (float*)alloc((size_t)H * 4);
    unsigned short* qbuf = (unsigned short*)alloc((size_t)M * H * 2);
    unsigned short* kbuf = (unsigned short*)alloc((size_t)M * H * 2);
    unsigned short* vbuf = (unsigned short*)alloc((size_t)M * H * 2);
    unsigned short* ctxb = (unsigned short*)alloc((size_t)M * H * 2);

    const int nx = M * H;
    const int nw = H * H;
    cvt_f32_bf16<<<(nx + 255) / 256, 256, 0, stream>>>(x,  xb,  nx);
    cvt_wq_scaled<<<(nw + 255) / 256, 256, 0, stream>>>(Wq, scale, wqb, nw);
    cvt_f32_bf16<<<(nw + 255) / 256, 256, 0, stream>>>(Wk, wkb, nw);
    cvt_f32_bf16<<<(nw + 255) / 256, 256, 0, stream>>>(Wv, wvb, nw);
    cvt_f32_bf16<<<(nw + 255) / 256, 256, 0, stream>>>(Wo, wob, nw);
    scale_bias<<<(H + 255) / 256, 256, 0, stream>>>(bq, scale, bqs, H);

    dim3 gg(H / BN, M / BM);   // (4, 128)
    gemm_bf16_nt<<<gg, 256, 0, stream>>>(xb, wqb, bqs, qbuf, M, H, H, 1);
    gemm_bf16_nt<<<gg, 256, 0, stream>>>(xb, wkb, bk,  kbuf, M, H, H, 1);
    gemm_bf16_nt<<<gg, 256, 0, stream>>>(xb, wvb, bv,  vbuf, M, H, H, 1);

    attn_kernel<<<M / TOK_PER_BLK, 128, 0, stream>>>(qbuf, kbuf, vbuf, ctxb);

    gemm_bf16_nt<<<gg, 256, 0, stream>>>(ctxb, wob, bo, d_out, M, H, H, 0);
}
